// GCN_5Layers_21388937134412
// MI455X (gfx1250) — matrix operand order repaired and checked by reading, not compiled, not run on hardware
//
#include <hip/hip_runtime.h>
#include <hip/hip_bf16.h>

// ---------------------------------------------------------------------------
// Types for CDNA5 WMMA
// ---------------------------------------------------------------------------
typedef __attribute__((ext_vector_type(16))) __bf16 v16bf;
typedef __attribute__((ext_vector_type(8)))  __bf16 v8bf;
typedef __attribute__((ext_vector_type(4)))  __bf16 v4bf;
typedef __attribute__((ext_vector_type(8)))  float  v8f;

#define F_IN 128   // K dimension for every layer

// Native non-returning f32 atomic add (global_atomic_add_f32).
__device__ __forceinline__ void atomic_add_f32(float* p, float v) {
    unsafeAtomicAdd(p, v);
}

// ---------------------------------------------------------------------------
// Utility kernels
// ---------------------------------------------------------------------------
__global__ __launch_bounds__(256) void k_zero_f4(float4* __restrict__ p, long long n4) {
    long long t = (long long)blockIdx.x * blockDim.x + threadIdx.x;
    long long stride = (long long)gridDim.x * blockDim.x;
    float4 z = {0.f, 0.f, 0.f, 0.f};
    for (long long i = t; i < n4; i += stride) p[i] = z;
}

__global__ __launch_bounds__(256) void k_copy_f4(const float4* __restrict__ s,
                                                 float4* __restrict__ d, long long n4) {
    long long t = (long long)blockIdx.x * blockDim.x + threadIdx.x;
    if (t < n4) d[t] = s[t];
}

// deg_out[src[e]] += 1 ; deg_in[dst[e]] += 1   (deg buffers contiguous)
__global__ __launch_bounds__(256) void k_degrees(const int* __restrict__ src,
                                                 const int* __restrict__ dst,
                                                 float* __restrict__ deg_out,
                                                 float* __restrict__ deg_in,
                                                 long long E) {
    long long e = (long long)blockIdx.x * blockDim.x + threadIdx.x;
    if (e >= E) return;
    atomic_add_f32(deg_out + src[e], 1.0f);
    atomic_add_f32(deg_in  + dst[e], 1.0f);
}

// inv = deg > 0 ? rsqrt(deg) : 0      (processes 2N entries: out then in)
__global__ __launch_bounds__(256) void k_inv_sqrt(const float* __restrict__ deg,
                                                  float* __restrict__ inv, int n) {
    int t = blockIdx.x * blockDim.x + threadIdx.x;
    if (t >= n) return;
    float d = deg[t];
    inv[t] = (d > 0.0f) ? rsqrtf(d) : 0.0f;
}

// W [K][Ncol] f32  ->  Wt [Ncol][K] bf16   (tiny; once per launch per layer)
__global__ __launch_bounds__(256) void k_wt_cvt(const float* __restrict__ W,
                                                __bf16* __restrict__ Wt,
                                                int K, int Ncol) {
    int t = blockIdx.x * blockDim.x + threadIdx.x;
    if (t >= K * Ncol) return;
    int k = t / Ncol, n = t % Ncol;
    Wt[(size_t)n * K + k] = (__bf16)W[t];
}

// ---------------------------------------------------------------------------
// SpMM scatter:  agg[dst] += h[src] * inv_out[src]
// One wave (32 lanes) per edge per iteration, grid-stride over edges.
// 512B coalesced row gather, native f32 atomics into the dst row (L2-resident:
// the whole 51MB feature matrix fits in the 192MB L2).
// Prefetch (global_prefetch_b8) of the next iteration's source row overlaps
// the atomic drain with the next gather's tag/translation work.
// ---------------------------------------------------------------------------
__global__ __launch_bounds__(256) void k_scatter(const float* __restrict__ h,
                                                 const int* __restrict__ src,
                                                 const int* __restrict__ dst,
                                                 const float* __restrict__ inv_out,
                                                 float* __restrict__ agg,
                                                 long long E) {
    const int  lane  = threadIdx.x & 31;
    long long  wave  = ((long long)blockIdx.x * blockDim.x + threadIdx.x) >> 5;
    const long long nwaves = ((long long)gridDim.x * blockDim.x) >> 5;

    for (long long e = wave; e < E; e += nwaves) {
        int s = src[e];
        int d = dst[e];
        long long en = e + nwaves;
        if (en < E) {
            // warm L2/L0 for the next gather row while this edge's atomics drain
            __builtin_prefetch(h + (size_t)src[en] * F_IN + lane * 4, 0, 3);
        }
        float sc = inv_out[s];
        const float4 v = *(const float4*)(h + (size_t)s * F_IN + lane * 4);
        float* ap = agg + (size_t)d * F_IN + lane * 4;
        atomic_add_f32(ap + 0, v.x * sc);
        atomic_add_f32(ap + 1, v.y * sc);
        atomic_add_f32(ap + 2, v.z * sc);
        atomic_add_f32(ap + 3, v.w * sc);
    }
}

// ---------------------------------------------------------------------------
// A-operand prep:  abf[n][f] = bf16( agg[n][f] * inv_in[n] )
// Thread handles 4 features; single 8B store.
// ---------------------------------------------------------------------------
__global__ __launch_bounds__(256) void k_scale_cvt(const float* __restrict__ agg,
                                                   const float* __restrict__ inv_in,
                                                   __bf16* __restrict__ abf, int N) {
    int t = blockIdx.x * blockDim.x + threadIdx.x;   // one per 4 features
    if (t >= N * (F_IN / 4)) return;
    int n = t >> 5;                                   // 32 quads per 128-wide row
    float sc = inv_in[n];
    const float4 v = *(const float4*)(agg + (size_t)t * 4);
    v4bf o;
    o.x = (__bf16)(v.x * sc);
    o.y = (__bf16)(v.y * sc);
    o.z = (__bf16)(v.z * sc);
    o.w = (__bf16)(v.w * sc);
    *(v4bf*)(abf + (size_t)t * 4) = o;
}

// ---------------------------------------------------------------------------
// WMMA GEMM:  out[M][NCOLS] = act( A[M][128](bf16) @ Wt^T + bias )
// One wave per 16x16 output tile; K=128 -> 4 x v_wmma_f32_16x16x32_bf16.
// M is guaranteed a multiple of 16 by the caller (N=100000), so the store
// epilogue is branch-free: one base address + 8 immediate-offset b32 stores.
//
// A per-lane layout (ISA 16-bit A 16x32): lanes 0-15 (row=lane) hold
//   K {0..7} in VGPR0-3 and {16..23} in VGPR4-7; lanes 16-31 hold +8.
// B per-lane layout (K x 16, lane = N col): same K order as A: lanes 0-15
//   hold K {0..7} and {16..23}; lanes 16-31 hold +8 (Wt is [N][K]).
// D layout: lane n = lane&15 fixed; VGPR v -> row m0 + v + (lane>=16 ? 8 : 0).
// ---------------------------------------------------------------------------
template <int NCOLS, bool RELU>
__global__ __launch_bounds__(32) void k_wmma_gemm(const __bf16* __restrict__ A,
                                                  const __bf16* __restrict__ Wt,
                                                  const float* __restrict__ bias,
                                                  float* __restrict__ out) {
    const int lane  = threadIdx.x;           // 0..31
    const int ln    = lane & 15;
    const int laneH = lane >> 4;             // 0 or 1
    const int m0    = blockIdx.x * 16;
    const int n0    = blockIdx.y * 16;

    const __bf16* arow = A  + (size_t)(m0 + ln) * F_IN + laneH * 8;
    const __bf16* brow = Wt + (size_t)(n0 + ln) * F_IN + laneH * 8;

    v8f c = {};
#pragma unroll
    for (int kk = 0; kk < F_IN; kk += 32) {
        v8bf a_lo = *(const v8bf*)(arow + kk);        // K {0..7}  (+8 hi lanes)
        v8bf a_hi = *(const v8bf*)(arow + kk + 16);   // K {16..23}(+8 hi lanes)
        v8bf b_lo = *(const v8bf*)(brow + kk);        // K {0..7}  (+8 hi lanes)
        v8bf b_hi = *(const v8bf*)(brow + kk + 16);   // K {16..23}(+8 hi lanes)
        v16bf a = __builtin_shufflevector(a_lo, a_hi, 0,1,2,3,4,5,6,7,
                                                      8,9,10,11,12,13,14,15);
        v16bf b = __builtin_shufflevector(b_lo, b_hi, 0,1,2,3,4,5,6,7,
                                                      8,9,10,11,12,13,14,15);
        c = __builtin_amdgcn_wmma_f32_16x16x32_bf16(false, a, false, b,
                                                    (short)0, c, false, false);
    }

    const float bn = bias[n0 + ln];
    float* orow = out + (size_t)(m0 + laneH * 8) * NCOLS + (n0 + ln);
#pragma unroll
    for (int v = 0; v < 8; ++v) {
        float r = c[v] + bn;
        if (RELU) r = fmaxf(r, 0.0f);
        orow[(size_t)v * NCOLS] = r;   // 8 stores, immediate offsets
    }
}

// ---------------------------------------------------------------------------
// Host orchestration
// ---------------------------------------------------------------------------
extern "C" void kernel_launch(void* const* d_in, const int* in_sizes, int n_in,
                              void* d_out, int out_size, void* d_ws, size_t ws_size,
                              hipStream_t stream) {
    const float* x   = (const float*)d_in[0];
    const int*   src = (const int*)  d_in[1];
    const int*   dst = (const int*)  d_in[2];
    const float* W[5]    = {(const float*)d_in[3], (const float*)d_in[5],
                            (const float*)d_in[7], (const float*)d_in[9],
                            (const float*)d_in[11]};
    const float* bias[5] = {(const float*)d_in[4], (const float*)d_in[6],
                            (const float*)d_in[8], (const float*)d_in[10],
                            (const float*)d_in[12]};

    const int       N     = in_sizes[0] / F_IN;      // 100000 (multiple of 16)
    const long long E     = in_sizes[1];             // 1.6M
    const int       out_f = in_sizes[12];            // 64
    const int       Mpad  = (N + 15) & ~15;

    // ---- workspace partition (256B aligned) ----
    char*  ws = (char*)d_ws;
    size_t o  = 0;
    auto take = [&](size_t bytes) { char* p = ws + o; o = (o + bytes + 255) & ~(size_t)255; return p; };
    float*  deg    = (float*) take(2 * (size_t)N * 4);            // deg_out | deg_in
    float*  inv    = (float*) take(2 * (size_t)N * 4);            // inv_out | inv_in
    float*  agg    = (float*) take((size_t)N * F_IN * 4);
    __bf16* abf    = (__bf16*)take((size_t)Mpad * F_IN * 2);
    float*  hA     = (float*) take((size_t)N * F_IN * 4);
    float*  hB     = (float*) take((size_t)N * F_IN * 4);
    __bf16* Wt[5];
    for (int i = 0; i < 5; ++i) Wt[i] = (__bf16*)take((size_t)F_IN * F_IN * 2);
    (void)ws_size;

    float* out_h5 = (float*)d_out;                                 // [N][64]
    float* out_h3 = (float*)d_out + (size_t)N * out_f;             // [N][128]
    float* out_x  = out_h3 + (size_t)N * F_IN;                     // [N][128]

    const int TB = 256;
    auto cdiv = [](long long a, long long b) { return (int)((a + b - 1) / b); };

    // ---- degrees + norms ----
    k_zero_f4<<<2048, TB, 0, stream>>>((float4*)deg, (long long)(2 * (size_t)N) / 4);
    k_degrees<<<cdiv(E, TB), TB, 0, stream>>>(src, dst, deg, deg + N, E);
    k_inv_sqrt<<<cdiv(2 * (long long)N, TB), TB, 0, stream>>>(deg, inv, 2 * N);
    const float* inv_out = inv;
    const float* inv_in  = inv + N;

    // ---- weight transpose + bf16 convert ----
    for (int i = 0; i < 5; ++i) {
        int nc = (i == 4) ? out_f : F_IN;
        k_wt_cvt<<<cdiv((long long)F_IN * nc, TB), TB, 0, stream>>>(W[i], Wt[i], F_IN, nc);
    }

    // Grid-stride scatter: ~400k waves, each handling ~4 edges (gives the
    // prefetch a target while staying fully occupancy-saturating).
    const int scatter_blocks = 50000;

    // ---- one GCN layer ----
    auto layer = [&](const float* hin, int wi, float* hout, int ncols, bool relu) {
        k_zero_f4<<<4096, TB, 0, stream>>>((float4*)agg, (long long)N * F_IN / 4);
        k_scatter<<<scatter_blocks, TB, 0, stream>>>(hin, src, dst, inv_out, agg, E);
        k_scale_cvt<<<cdiv((long long)N * (F_IN / 4), TB), TB, 0, stream>>>(agg, inv_in, abf, N);
        dim3 grid(Mpad / 16, ncols / 16);
        if (ncols == F_IN) {
            if (relu) k_wmma_gemm<F_IN, true ><<<grid, 32, 0, stream>>>(abf, Wt[wi], bias[wi], hout);
            else      k_wmma_gemm<F_IN, false><<<grid, 32, 0, stream>>>(abf, Wt[wi], bias[wi], hout);
        } else {
            k_wmma_gemm<64, false><<<grid, 32, 0, stream>>>(abf, Wt[wi], bias[wi], hout);
        }
    };

    layer(x,      0, hA,     F_IN, true);   // h1
    layer(hA,     1, hB,     F_IN, true);   // h2
    layer(hB,     2, out_h3, F_IN, true);   // h3 -> straight into d_out slice
    layer(out_h3, 3, hA,     F_IN, true);   // h4
    layer(hA,     4, out_h5, 64,   false);  // h5 -> d_out[0:]

    // ---- x passthrough output ----
    k_copy_f4<<<cdiv((long long)N * F_IN / 4, TB), TB, 0, stream>>>(
        (const float4*)x, (float4*)out_x, (long long)N * F_IN / 4);
}
